// T2SLayer_73349451481740
// MI455X (gfx1250) — compile-verified, matrix operand order checked by reading, not run on hardware
//
#include <hip/hip_runtime.h>
#include <math.h>

// ---------------- problem constants ----------------
#define DMODEL 512
#define NHEAD  8
#define EHEAD  64
#define BATCH  8
#define LLEN   12
#define NNODE  207
#define SEQT   (LLEN * NNODE)          // 2484
#define BT_ROWS (BATCH * SEQT)         // 19872
#define FFN    2048
#define OUTL   12
#define CCATC  4608                    // D*(NG+1)*ORDER + D

typedef __attribute__((ext_vector_type(16))) _Float16 v16h;
typedef __attribute__((ext_vector_type(8)))  _Float16 v8h;
typedef __attribute__((ext_vector_type(8)))  float    v8f;
typedef __attribute__((ext_vector_type(4)))  float    v4f;

union AFrag { v16h v; v8h h2[2]; };
union VFrag { v8f v; float f[8]; };
union PackH { v8h v; _Float16 h[8]; };
union F4    { v4f v; float f[4]; };

__device__ __forceinline__ float act_apply(float x, int act) {
  if (act == 1) {            // gelu (tanh approx, jax default)
    float x3 = x * x * x;
    return 0.5f * x * (1.0f + tanhf(0.79788456080286535f * (x + 0.044715f * x3)));
  } else if (act == 2) {     // elu + 1
    return x > 0.0f ? x + 1.0f : __expf(x);
  }
  return x;
}

// ---------------------------------------------------------------------------
// FAST GEMM: C = act(A (MxK) @ W (KxN) + bias) (+res).  Preconditions:
//   K % 32 == 0, Nc % 64 == 0, lda == K, ldw == Nc, A/W/C 16B-aligned.
// M may be ragged: A row index clamped (in-bounds), stores guarded by row<M.
// LDS double-buffered, register-staged pipeline, branch-free K-loop.
// ---------------------------------------------------------------------------
__global__ __launch_bounds__(256) void wmma_gemm_fast_kernel(
    const float* __restrict__ A, const float* __restrict__ W,
    const float* __restrict__ bias, const float* __restrict__ res,
    float* __restrict__ C, int M, int Nc, int K, int act)
{
  __shared__ __align__(16) _Float16 As[2][64][40];   // 80B row stride
  __shared__ __align__(16) _Float16 Bs[2][64][40];   // Bs[n][k]

  int tid  = threadIdx.x;
  int wave = tid >> 5, lane = tid & 31;
  int wm = wave & 3;        // M sub-tile (16 rows)
  int wn = wave >> 2;       // N 32-col block
  int m0 = blockIdx.y * 64, n0 = blockIdx.x * 64;

  v8f zero = {};
  VFrag acc0, acc1; acc0.v = zero; acc1.v = zero;

  int lrow = tid >> 2;              // 0..63 : A LDS row this thread fills
  int kc   = (tid & 3) << 3;        // 0,8,16,24
  int ar   = wm * 16 + (lane & 15);
  int br0  = wn * 32 + (lane & 15);
  int br1  = br0 + 16;
  int kgrp = (lane >> 4) << 3;      // lane K group: 0 or 8
  int bkk  = tid >> 3;              // 0..31 : W row (k) this thread loads
  int bnc  = (tid & 7) << 3;        // n group (8 consecutive n)

  int arow = m0 + lrow; if (arow >= M) arow = M - 1;   // clamp once; in-bounds
  const float* aptr = A + (long)arow * K + kc;
  const float* wptr = W + (long)bkk * Nc + n0 + bnc;

  F4 a0, a1, b0, b1;                // register-staged tile
  a0.v = *(const v4f*)aptr;  a1.v = *(const v4f*)(aptr + 4);
  b0.v = *(const v4f*)wptr;  b1.v = *(const v4f*)(wptr + 4);

  auto stage = [&](int buf) {
    PackH pa;
    #pragma unroll
    for (int j = 0; j < 4; ++j) {
      pa.h[j]     = (_Float16)a0.f[j];
      pa.h[4 + j] = (_Float16)a1.f[j];
    }
    *(v8h*)&As[buf][lrow][kc] = pa.v;
    #pragma unroll
    for (int j = 0; j < 4; ++j) {
      Bs[buf][bnc + j][bkk]     = (_Float16)b0.f[j];
      Bs[buf][bnc + 4 + j][bkk] = (_Float16)b1.f[j];
    }
  };

  int nk = K >> 5;
  stage(0);
  __syncthreads();

  for (int it = 0; it < nk; ++it) {
    int cur = it & 1;
    bool more = (it + 1) < nk;
    if (more) {                                  // prefetch next tile (overlaps WMMA)
      aptr += 32; wptr += (long)32 * Nc;
      a0.v = *(const v4f*)aptr; a1.v = *(const v4f*)(aptr + 4);
      b0.v = *(const v4f*)wptr; b1.v = *(const v4f*)(wptr + 4);
    }

    AFrag af, bf0, bf1;
    af.h2[0]  = *(const v8h*)&As[cur][ar][kgrp];
    af.h2[1]  = *(const v8h*)&As[cur][ar][kgrp + 16];
    bf0.h2[0] = *(const v8h*)&Bs[cur][br0][kgrp];
    bf0.h2[1] = *(const v8h*)&Bs[cur][br0][kgrp + 16];
    bf1.h2[0] = *(const v8h*)&Bs[cur][br1][kgrp];
    bf1.h2[1] = *(const v8h*)&Bs[cur][br1][kgrp + 16];

    acc0.v = __builtin_amdgcn_wmma_f32_16x16x32_f16(
        false, af.v, false, bf0.v, (short)0, acc0.v, false, false);
    acc1.v = __builtin_amdgcn_wmma_f32_16x16x32_f16(
        false, af.v, false, bf1.v, (short)0, acc1.v, false, false);

    if (more) stage(cur ^ 1);
    __syncthreads();                             // single barrier per K-step
  }

  int crow  = m0 + wm * 16 + ((lane >> 4) << 3);
  int ccol0 = n0 + wn * 32 + (lane & 15);
  int ccol1 = ccol0 + 16;
  float bs0 = bias ? bias[ccol0] : 0.0f;
  float bs1 = bias ? bias[ccol1] : 0.0f;
  #pragma unroll
  for (int r = 0; r < 8; ++r) {
    int row = crow + r;
    if (row >= M) continue;
    float v0 = act_apply(acc0.f[r] + bs0, act);
    float v1 = act_apply(acc1.f[r] + bs1, act);
    if (res) {
      v0 += res[(long)row * Nc + ccol0];
      v1 += res[(long)row * Nc + ccol1];
    }
    C[(long)row * Nc + ccol0] = v0;
    C[(long)row * Nc + ccol1] = v1;
  }
}

// ---------------------------------------------------------------------------
// Generic batched GEMM: C[b,o] = act(A[b,o] (MxK) @ W[b,o] (KxN) + bias) (+res)
// wt=1 -> W stored as (N x K) row-major. Edge tiles: clamp-and-select loads.
// ---------------------------------------------------------------------------
__global__ __launch_bounds__(256) void wmma_gemm_kernel(
    const float* __restrict__ A, const float* __restrict__ W,
    const float* __restrict__ bias, const float* __restrict__ res,
    float* __restrict__ C,
    int M, int Nc, int K, int lda, int ldw, int ldc,
    long a_sb, long a_so, long w_sb, long w_so, long c_sb, long c_so,
    int Odim, int wt, int act)
{
  int bo = blockIdx.z;
  int bb = bo / Odim, oo = bo % Odim;
  A += (long)bb * a_sb + (long)oo * a_so;
  W += (long)bb * w_sb + (long)oo * w_so;
  C += (long)bb * c_sb + (long)oo * c_so;
  if (res) res += (long)bb * c_sb + (long)oo * c_so;

  __shared__ __align__(16) _Float16 As[2][64][40];
  __shared__ __align__(16) _Float16 Bs[2][64][40];

  int tid  = threadIdx.x;
  int wave = tid >> 5, lane = tid & 31;
  int wm = wave & 3;
  int wn = wave >> 2;
  int m0 = blockIdx.y * 64, n0 = blockIdx.x * 64;

  v8f zero = {};
  VFrag acc0, acc1; acc0.v = zero; acc1.v = zero;

  int lrow = tid >> 2;
  int kc   = (tid & 3) << 3;
  int ar   = wm * 16 + (lane & 15);
  int br0  = wn * 32 + (lane & 15);
  int br1  = br0 + 16;
  int kgrp = (lane >> 4) << 3;

  float av[8], bv[8];

  auto load_tile = [&](int k0) {
    int gr = m0 + lrow;
    int cr = gr < M ? gr : M - 1;
    bool rok = gr < M;
    const float* arow = A + (long)cr * lda;
    #pragma unroll
    for (int j = 0; j < 8; ++j) {
      int gk = k0 + kc + j;
      int ck2 = gk < K ? gk : K - 1;
      float v = arow[ck2];                       // always in-bounds (clamped)
      av[j] = (rok && gk < K) ? v : 0.0f;        // branch-free select
    }
    int gn = n0 + lrow;
    int cn = gn < Nc ? gn : Nc - 1;
    bool nok = gn < Nc;
    if (wt) {
      const float* wrow = W + (long)cn * ldw;
      #pragma unroll
      for (int j = 0; j < 8; ++j) {
        int gk = k0 + kc + j;
        int ck2 = gk < K ? gk : K - 1;
        float v = wrow[ck2];
        bv[j] = (nok && gk < K) ? v : 0.0f;
      }
    } else {
      #pragma unroll
      for (int j = 0; j < 8; ++j) {
        int gk = k0 + kc + j;
        int ck2 = gk < K ? gk : K - 1;
        float v = W[(long)ck2 * ldw + cn];
        bv[j] = (nok && gk < K) ? v : 0.0f;
      }
    }
  };

  auto store_tile = [&](int buf) {
    #pragma unroll
    for (int j = 0; j < 8; ++j) As[buf][lrow][kc + j] = (_Float16)av[j];
    #pragma unroll
    for (int j = 0; j < 8; ++j) Bs[buf][lrow][kc + j] = (_Float16)bv[j];
  };

  int nk = (K + 31) >> 5;
  load_tile(0);
  store_tile(0);
  __syncthreads();

  for (int it = 0; it < nk; ++it) {
    int cur = it & 1;
    bool more = (it + 1) < nk;
    if (more) load_tile((it + 1) << 5);

    AFrag af, bf0, bf1;
    af.h2[0]  = *(const v8h*)&As[cur][ar][kgrp];
    af.h2[1]  = *(const v8h*)&As[cur][ar][kgrp + 16];
    bf0.h2[0] = *(const v8h*)&Bs[cur][br0][kgrp];
    bf0.h2[1] = *(const v8h*)&Bs[cur][br0][kgrp + 16];
    bf1.h2[0] = *(const v8h*)&Bs[cur][br1][kgrp];
    bf1.h2[1] = *(const v8h*)&Bs[cur][br1][kgrp + 16];

    acc0.v = __builtin_amdgcn_wmma_f32_16x16x32_f16(
        false, af.v, false, bf0.v, (short)0, acc0.v, false, false);
    acc1.v = __builtin_amdgcn_wmma_f32_16x16x32_f16(
        false, af.v, false, bf1.v, (short)0, acc1.v, false, false);

    if (more) store_tile(cur ^ 1);
    __syncthreads();
  }

  int crow  = m0 + wm * 16 + ((lane >> 4) << 3);
  int ccol0 = n0 + wn * 32 + (lane & 15);
  int ccol1 = ccol0 + 16;
  #pragma unroll
  for (int r = 0; r < 8; ++r) {
    int row = crow + r;
    if (row >= M) continue;
    if (ccol0 < Nc) {
      float v = acc0.f[r] + (bias ? bias[ccol0] : 0.0f);
      v = act_apply(v, act);
      if (res) v += res[(long)row * ldc + ccol0];
      C[(long)row * ldc + ccol0] = v;
    }
    if (ccol1 < Nc) {
      float v = acc1.f[r] + (bias ? bias[ccol1] : 0.0f);
      v = act_apply(v, act);
      if (res) v += res[(long)row * ldc + ccol1];
      C[(long)row * ldc + ccol1] = v;
    }
  }
}

// ---------------------------------------------------------------------------
// kv[b,h,e,d] = sum_t kf[b,t,h,e] * v[b,t,h,d];  ksum[b,h,e] = sum_t kf
// ---------------------------------------------------------------------------
__global__ __launch_bounds__(256) void kv_ksum_kernel(
    const float* __restrict__ kf, const float* __restrict__ vv,
    float* __restrict__ kv, float* __restrict__ ksum)
{
  int bh = blockIdx.x;
  int b = bh >> 3, h = bh & 7;
  const float* kb = kf + (long)b * SEQT * DMODEL + h * EHEAD;
  const float* vb = vv + (long)b * SEQT * DMODEL + h * EHEAD;

  __shared__ float ks[16][EHEAD];
  __shared__ float vs[16][EHEAD];

  int tid = threadIdx.x;
  int e  = tid >> 2;
  int dg = (tid & 3) << 4;

  float acc[16];
  #pragma unroll
  for (int i = 0; i < 16; ++i) acc[i] = 0.0f;
  float ksacc = 0.0f;

  for (int t0 = 0; t0 < SEQT; t0 += 16) {
    int nt = SEQT - t0; if (nt > 16) nt = 16;
    for (int i = tid; i < 16 * EHEAD; i += 256) {
      int tt = i >> 6, dd = i & 63;
      float kl = 0.0f, vl = 0.0f;
      if (tt < nt) {
        kl = kb[(long)(t0 + tt) * DMODEL + dd];
        vl = vb[(long)(t0 + tt) * DMODEL + dd];
      }
      ks[tt][dd] = kl; vs[tt][dd] = vl;
    }
    __syncthreads();
    for (int tt = 0; tt < nt; ++tt) {
      float kval = ks[tt][e];
      if ((tid & 3) == 0) ksacc += kval;
      #pragma unroll
      for (int i = 0; i < 16; ++i) acc[i] += kval * vs[tt][dg + i];
    }
    __syncthreads();
  }

  float* kvo = kv + (long)bh * EHEAD * EHEAD;
  #pragma unroll
  for (int i = 0; i < 16; ++i) kvo[e * EHEAD + dg + i] = acc[i];
  if ((tid & 3) == 0) ksum[bh * EHEAD + e] = ksacc;
}

// ---------------------------------------------------------------------------
// o[b,t,h,d] = (sum_e qf*kv[b,h,e,d]) / (sum_e qf*ksum[b,h,e] + EPS)
// ---------------------------------------------------------------------------
__global__ __launch_bounds__(256) void attn_o_kernel(
    const float* __restrict__ qf, const float* __restrict__ kv,
    const float* __restrict__ ksum, float* __restrict__ o)
{
  long row = blockIdx.x;
  int b = (int)(row / SEQT);
  const float* qr = qf + row * DMODEL;

  __shared__ float qs[DMODEL];
  __shared__ float kss[DMODEL];
  int tid = threadIdx.x;
  for (int i = tid; i < DMODEL; i += 256) {
    qs[i]  = qr[i];
    kss[i] = ksum[b * DMODEL + i];
  }
  __syncthreads();

  int d = tid & 63;
  for (int h = tid >> 6; h < NHEAD; h += 4) {
    const float* qh = &qs[h * EHEAD];
    const float* kh = &kss[h * EHEAD];
    const float* kvb = kv + ((long)(b * NHEAD + h)) * EHEAD * EHEAD + d;
    float dot = 0.0f, oacc = 0.0f;
    #pragma unroll 8
    for (int e = 0; e < EHEAD; ++e) {
      float qv = qh[e];
      dot  += qv * kh[e];
      oacc += qv * kvb[e * EHEAD];
    }
    o[row * DMODEL + h * EHEAD + d] = oacc / (dot + 1e-6f);
  }
}

// ---------------------------------------------------------------------------
// LayerNorm over last dim (512), y = (x-mean)/sqrt(var+1e-5)*g + b
// ---------------------------------------------------------------------------
__global__ __launch_bounds__(256) void ln_kernel(
    const float* __restrict__ x, const float* __restrict__ g,
    const float* __restrict__ bt, float* __restrict__ y)
{
  long row = blockIdx.x;
  const float* xr = x + row * DMODEL;
  float* yr = y + row * DMODEL;
  int tid = threadIdx.x;

  float a0 = xr[tid], a1 = xr[tid + 256];
  __shared__ float red[256];
  red[tid] = a0 + a1;
  __syncthreads();
  for (int off = 128; off > 0; off >>= 1) {
    if (tid < off) red[tid] += red[tid + off];
    __syncthreads();
  }
  float mean = red[0] * (1.0f / 512.0f);
  __syncthreads();
  float d0 = a0 - mean, d1 = a1 - mean;
  red[tid] = d0 * d0 + d1 * d1;
  __syncthreads();
  for (int off = 128; off > 0; off >>= 1) {
    if (tid < off) red[tid] += red[tid + off];
    __syncthreads();
  }
  float var = red[0] * (1.0f / 512.0f);
  float rstd = rsqrtf(var + 1e-5f);
  yr[tid]       = d0 * rstd * g[tid]       + bt[tid];
  yr[tid + 256] = d1 * rstd * g[tid + 256] + bt[tid + 256];
}

// ---------------------------------------------------------------------------
// xg[b,o,d,n] = sum_l tt_w[o,l]*t_out[b,l,n,d] + tt_b[o]  (into cat segment 0)
// ---------------------------------------------------------------------------
__global__ __launch_bounds__(256) void tt_kernel(
    const float* __restrict__ t_out, const float* __restrict__ tt_w,
    const float* __restrict__ tt_b, float* __restrict__ cat)
{
  int bo = blockIdx.z;
  int b = bo / OUTL, o = bo % OUTL;
  int idx = blockIdx.x * 256 + threadIdx.x;
  int d = idx & 511;
  int n = idx >> 9;
  float acc = tt_b[o];
  #pragma unroll
  for (int l = 0; l < LLEN; ++l)
    acc += tt_w[o * LLEN + l] * t_out[(((long)b * LLEN + l) * NNODE + n) * DMODEL + d];
  cat[((long)bo * CCATC + d) * NNODE + n] = acc;
}

// adjs[s][n][m] = adj_mats[n,m,s]
__global__ __launch_bounds__(256) void adj_repack_kernel(
    const float* __restrict__ adj, float* __restrict__ adjs)
{
  int i = blockIdx.x * 256 + threadIdx.x;
  if (i >= 3 * NNODE * NNODE) return;
  int s = i / (NNODE * NNODE);
  int r = i % (NNODE * NNODE);
  adjs[i] = adj[r * 3 + s];
}

// dts[b,o,n,d] = (g[b,o,d,n] + gcn_b[o,d]) * bn_g[o,d] + bn_b[o,d]
__global__ __launch_bounds__(256) void bn_tr_kernel(
    const float* __restrict__ gmat, const float* __restrict__ gcn_b,
    const float* __restrict__ bn_g, const float* __restrict__ bn_b,
    float* __restrict__ dts)
{
  int bo = blockIdx.z;
  int o = bo % OUTL;
  int idx = blockIdx.x * 256 + threadIdx.x;
  int n = idx % NNODE;
  int d = idx / NNODE;
  float v = gmat[((long)bo * DMODEL + d) * NNODE + n];
  int od = o * DMODEL + d;
  v = (v + gcn_b[od]) * bn_g[od] + bn_b[od];
  dts[((long)bo * NNODE + n) * DMODEL + d] = v;
}

// ---------------------------------------------------------------------------
extern "C" void kernel_launch(void* const* d_in, const int* in_sizes, int n_in,
                              void* d_out, int out_size, void* d_ws, size_t ws_size,
                              hipStream_t stream) {
  (void)in_sizes; (void)n_in; (void)out_size; (void)ws_size;

  const float* x    = (const float*)d_in[0];
  const float* st   = (const float*)d_in[1];
  const float* adj  = (const float*)d_in[3];
  const float* eqw  = (const float*)d_in[4];
  const float* eqb  = (const float*)d_in[5];
  const float* ew1  = (const float*)d_in[6];
  const float* eb1  = (const float*)d_in[7];
  const float* ew2  = (const float*)d_in[8];
  const float* eb2  = (const float*)d_in[9];
  const float* elng = (const float*)d_in[10];
  const float* elnb = (const float*)d_in[11];
  const float* dsw  = (const float*)d_in[12];
  const float* dsb  = (const float*)d_in[13];
  const float* dcw  = (const float*)d_in[14];
  const float* dcb  = (const float*)d_in[15];
  const float* dw1  = (const float*)d_in[16];
  const float* db1  = (const float*)d_in[17];
  const float* dw2  = (const float*)d_in[18];
  const float* db2  = (const float*)d_in[19];
  const float* dlng = (const float*)d_in[20];
  const float* dlnb = (const float*)d_in[21];
  const float* ttw  = (const float*)d_in[22];
  const float* ttb  = (const float*)d_in[23];
  const float* tsw  = (const float*)d_in[24];
  const float* tsb  = (const float*)d_in[25];
  const float* gw   = (const float*)d_in[26];
  const float* gb   = (const float*)d_in[27];
  const float* bng  = (const float*)d_in[28];
  const float* bnb  = (const float*)d_in[29];

  float* ws = (float*)d_ws;
  const long SZ = (long)BT_ROWS * DMODEL;
  float* bufQ   = ws;
  float* bufK   = bufQ + SZ;
  float* bufV   = bufK + SZ;
  float* bufO   = bufV + SZ;
  float* bufH   = bufO + SZ;
  float* bufY   = bufH + SZ;
  float* bufMEM = bufY + SZ;
  float* bufQ2  = bufMEM + SZ;
  float* bufF   = bufQ2 + SZ;                                   // BT x 2048
  float* bufS2  = bufF + (long)BT_ROWS * FFN;                   // B,L,N,N
  float* bufG   = bufS2 + (long)BATCH * LLEN * NNODE * NNODE;   // B,12,512,207
  float* bufCAT = bufG + (long)BATCH * OUTL * DMODEL * NNODE;   // B,12,4608,207
  float* kvb    = bufCAT + (long)BATCH * OUTL * CCATC * NNODE;
  float* ksb    = kvb + BATCH * NHEAD * EHEAD * EHEAD;
  float* adjs   = ksb + BATCH * NHEAD * EHEAD;

  float* dts   = (float*)d_out;
  float* s_out = (float*)d_out + (long)BATCH * OUTL * NNODE * DMODEL;

  // heavy GEMMs: K%32==0, Nc%64==0, wt=0 -> fast branch-free kernel
  auto gemmF = [&](const float* A, const float* Wm, const float* bias,
                   const float* rp, float* Cm, long M, int Nc, int K, int act) {
    dim3 g(Nc / 64, (unsigned)((M + 63) / 64), 1);
    wmma_gemm_fast_kernel<<<g, dim3(256), 0, stream>>>(
        A, Wm, bias, rp, Cm, (int)M, Nc, K, act);
  };

  auto gemm2d = [&](const float* A, const float* Wm, const float* bias,
                    const float* rp, float* Cm, long M, int Nc, int K,
                    int wtf, int act) {
    dim3 g((Nc + 63) / 64, (unsigned)((M + 63) / 64), 1);
    wmma_gemm_kernel<<<g, dim3(256), 0, stream>>>(
        A, Wm, bias, rp, Cm, (int)M, Nc, K,
        K, wtf ? K : Nc, Nc, 0L, 0L, 0L, 0L, 0L, 0L, 1, wtf, act);
  };

  auto attn_block = [&](const float* xq, const float* xkv, const float* w,
                        const float* bi, const float* resid, float* outH) {
    gemmF(xq,  w + 0L * DMODEL * DMODEL, bi + 0 * DMODEL, nullptr, bufQ, BT_ROWS, DMODEL, DMODEL, 2);
    gemmF(xkv, w + 1L * DMODEL * DMODEL, bi + 1 * DMODEL, nullptr, bufK, BT_ROWS, DMODEL, DMODEL, 2);
    gemmF(xkv, w + 2L * DMODEL * DMODEL, bi + 2 * DMODEL, nullptr, bufV, BT_ROWS, DMODEL, DMODEL, 0);
    kv_ksum_kernel<<<BATCH * NHEAD, dim3(256), 0, stream>>>(bufK, bufV, kvb, ksb);
    attn_o_kernel<<<BT_ROWS, dim3(256), 0, stream>>>(bufQ, kvb, ksb, bufO);
    gemmF(bufO, w + 3L * DMODEL * DMODEL, bi + 3 * DMODEL, resid, outH, BT_ROWS, DMODEL, DMODEL, 0);
  };

  // ---------------- encoder ----------------
  attn_block(x, x, eqw, eqb, x, bufH);
  ln_kernel<<<BT_ROWS, dim3(256), 0, stream>>>(bufH, elng, elnb, bufY);
  gemmF(bufY, ew1, eb1, nullptr, bufF, BT_ROWS, FFN, DMODEL, 1);
  gemmF(bufF, ew2, eb2, bufY, bufH, BT_ROWS, DMODEL, FFN, 0);
  ln_kernel<<<BT_ROWS, dim3(256), 0, stream>>>(bufH, elng + DMODEL, elnb + DMODEL, bufMEM);

  // ---------------- decoder self ----------------
  attn_block(st, st, dsw, dsb, st, bufH);
  ln_kernel<<<BT_ROWS, dim3(256), 0, stream>>>(bufH, dlng, dlnb, bufQ2);
  // cross attention (queries from bufQ2, kv from encoder memory)
  attn_block(bufQ2, bufMEM, dcw, dcb, bufQ2, bufH);
  ln_kernel<<<BT_ROWS, dim3(256), 0, stream>>>(bufH, dlng + DMODEL, dlnb + DMODEL, bufQ2);
  // decoder FFN
  gemmF(bufQ2, dw1, db1, nullptr, bufF, BT_ROWS, FFN, DMODEL, 1);
  gemmF(bufF, dw2, db2, bufQ2, bufH, BT_ROWS, DMODEL, FFN, 0);
  ln_kernel<<<BT_ROWS, dim3(256), 0, stream>>>(bufH, dlng + 2 * DMODEL, dlnb + 2 * DMODEL, s_out);

  // ---------------- spatial projection s_out2 = s_out @ ts_w^T + ts_b ----------------
  gemm2d(s_out, tsw, tsb, nullptr, bufS2, BT_ROWS, NNODE, DMODEL, 1, 0);

  // ---------------- temporal projection -> xg (cat segment 0) ----------------
  tt_kernel<<<dim3(414, 1, BATCH * OUTL), dim3(256), 0, stream>>>(bufMEM, ttw, ttb, bufCAT);

  // ---------------- adjacency repack ----------------
  adj_repack_kernel<<<(3 * NNODE * NNODE + 255) / 256, dim3(256), 0, stream>>>(adj, adjs);

  // ---------------- graph propagation (batched WMMA GEMMs per (b,o)) ----------------
  const long CAT_SB = (long)OUTL * CCATC * NNODE;   // per-b stride in cat
  const long CAT_SO = (long)CCATC * NNODE;          // per-o stride in cat
  auto prop = [&](const float* Aseg, const float* Wm, long w_sb, long w_so, float* Cseg) {
    dim3 g(4, 8, BATCH * OUTL);
    wmma_gemm_kernel<<<g, dim3(256), 0, stream>>>(
        Aseg, Wm, nullptr, nullptr, Cseg,
        DMODEL, NNODE, NNODE, NNODE, NNODE, NNODE,
        CAT_SB, CAT_SO, w_sb, w_so, CAT_SB, CAT_SO, OUTL, 0, 0);
  };
  const long SEGSZ = (long)DMODEL * NNODE;
  for (int s = 0; s < 3; ++s) {
    prop(bufCAT,                       adjs + (long)s * NNODE * NNODE, 0, 0, bufCAT + (1 + 2 * s) * SEGSZ);
    prop(bufCAT + (1 + 2 * s) * SEGSZ, adjs + (long)s * NNODE * NNODE, 0, 0, bufCAT + (2 + 2 * s) * SEGSZ);
  }
  // adaptive adjacency: W = s_out2[b, o] (NxN per slice)
  prop(bufCAT,             bufS2, (long)LLEN * NNODE * NNODE, (long)NNODE * NNODE, bufCAT + 7 * SEGSZ);
  prop(bufCAT + 7 * SEGSZ, bufS2, (long)LLEN * NNODE * NNODE, (long)NNODE * NNODE, bufCAT + 8 * SEGSZ);

  // ---------------- final GCN contraction: g = gcn_w[o] @ cat[b,o] ----------------
  {
    dim3 g(4, 8, BATCH * OUTL);
    wmma_gemm_kernel<<<g, dim3(256), 0, stream>>>(
        gw, bufCAT, nullptr, nullptr, bufG,
        DMODEL, NNODE, CCATC, CCATC, NNODE, NNODE,
        0L, (long)DMODEL * CCATC,
        CAT_SB, CAT_SO,
        (long)OUTL * DMODEL * NNODE, (long)DMODEL * NNODE,
        OUTL, 0, 0);
  }

  // ---------------- bias + batchnorm affine + transpose -> dts ----------------
  bn_tr_kernel<<<dim3(414, 1, BATCH * OUTL), dim3(256), 0, stream>>>(bufG, gb, bng, bnb, dts);
}
